// RAL_249108103293
// MI455X (gfx1250) — matrix operand order repaired and checked by reading, not compiled, not run on hardware
//
#include <hip/hip_runtime.h>
#include <hip/hip_bf16.h>

typedef __bf16 bf16;
typedef __attribute__((ext_vector_type(8)))  bf16  bf16x8;
typedef __attribute__((ext_vector_type(16))) bf16  bf16x16;
typedef __attribute__((ext_vector_type(8)))  float f32x8;

#define BATCH 4
#define CCH   192
#define HH    128     // full-res H = W
#define hh    64      // downsampled h = w
#define LL    4096    // 64*64 patch positions
#define KK    1728    // 192 * 3 * 3

// ---------------------------------------------------------------- K1: bilinear resize (align_corners), 128 -> 64
__global__ void k_resize(const float* __restrict__ fg, float* __restrict__ fgs) {
  int i = blockIdx.x * blockDim.x + threadIdx.x;
  const int total = BATCH * CCH * hh * hh;
  if (i >= total) return;
  int x = i & 63, y = (i >> 6) & 63, bc = i >> 12;
  const float sc = 127.0f / 63.0f;
  float fy = y * sc, fx = x * sc;
  int y0 = (int)floorf(fy), x0 = (int)floorf(fx);
  int y1 = min(y0 + 1, HH - 1), x1 = min(x0 + 1, HH - 1);
  float wy = fy - (float)y0, wx = fx - (float)x0;
  const float* p = fg + (size_t)bc * HH * HH;
  float v00 = p[y0 * HH + x0], v01 = p[y0 * HH + x1];
  float v10 = p[y1 * HH + x0], v11 = p[y1 * HH + x1];
  float top = v00 * (1.f - wy) + v10 * wy;
  float bot = v01 * (1.f - wy) + v11 * wy;
  fgs[i] = top * (1.f - wx) + bot * wx;
}

// ---------------------------------------------------------------- K2: fg patch rows P[b][l][k] (bf16) + 1/norm
__global__ __launch_bounds__(256) void k_patch(const float* __restrict__ fgs,
                                               bf16* __restrict__ P,
                                               float* __restrict__ invn) {
  const int l = blockIdx.x;      // 0..4095
  const int b = blockIdx.y;      // 0..3
  const int oy = l >> 6, ox = l & 63;
  const float* img = fgs + (size_t)b * CCH * hh * hh;
  bf16* row = P + ((size_t)b * LL + l) * KK;
  float ss = 0.f;
  for (int k = threadIdx.x; k < KK; k += 256) {
    int c = k / 9, r = k - c * 9, kh = r / 3, kw = r - kh * 3;
    int y = oy + kh - 1, x = ox + kw - 1;
    float v = 0.f;
    if ((unsigned)y < (unsigned)hh && (unsigned)x < (unsigned)hh)
      v = img[((size_t)c * hh + y) * hh + x];
    row[k] = (bf16)v;
    ss += v * v;
  }
  __shared__ float red[256];
  red[threadIdx.x] = ss; __syncthreads();
  for (int s = 128; s > 0; s >>= 1) {
    if (threadIdx.x < s) red[threadIdx.x] += red[threadIdx.x + s];
    __syncthreads();
  }
  if (threadIdx.x == 0) invn[b * LL + l] = 1.0f / fmaxf(sqrtf(red[0]), 1e-4f);
}

// ---------------------------------------------------------------- K3: repack bg patches: Bg[b][kh*4+kw][c][l] (bf16)
__global__ void k_bgpack(const float* __restrict__ bg, bf16* __restrict__ Bg) {
  size_t i = (size_t)blockIdx.x * blockDim.x + threadIdx.x;
  const size_t total = (size_t)BATCH * 16 * CCH * LL;
  if (i >= total) return;
  int l = (int)(i & (LL - 1));
  size_t t = i >> 12;
  int c = (int)(t % CCH); t /= CCH;
  int combo = (int)(t & 15);
  int b = (int)(t >> 4);
  int kh = combo >> 2, kw = combo & 3;
  int oy = l >> 6, ox = l & 63;
  int y = 2 * oy + kh - 1, x = 2 * ox + kw - 1;
  float v = 0.f;
  if ((unsigned)y < (unsigned)HH && (unsigned)x < (unsigned)HH)
    v = bg[(((size_t)b * CCH + c) * HH + y) * HH + x];
  Bg[i] = (bf16)v;
}

// ---------------------------------------------------------------- K3b: zero the dummy attention row used for OOB B-tiles
__global__ void k_zerorow(bf16* __restrict__ zrow) {
  int i = blockIdx.x * blockDim.x + threadIdx.x;
  if (i < LL) zrow[i] = (bf16)0.0f;
}

// A fragment (16x32 bf16, row-major source): lane<16 -> K {0..15}; lane>=16 -> K {16..31}
__device__ __forceinline__ bf16x16 load_a_frag(const bf16* rowp, int k0, int hi) {
  bf16x8 loA = *(const bf16x8*)(rowp + k0 + (hi ? 16 : 0));
  bf16x8 hiA = *(const bf16x8*)(rowp + k0 + 8 + (hi ? 16 : 0));
  return __builtin_shufflevector(loA, hiA, 0, 1, 2, 3, 4, 5, 6, 7,
                                 8, 9, 10, 11, 12, 13, 14, 15);
}

#define WMMA_BF16(A, B, C) \
  __builtin_amdgcn_wmma_f32_16x16x32_bf16(false, (A), false, (B), (short)0, (C), false, false)

// ---------------------------------------------------------------- K4: Gram GEMM  scoreT[p][l] = invn[l] * <P_l, P_p>
// Register-blocked: each wave computes a 32(l) x 32(p) block = 2x2 WMMA tiles.
__global__ __launch_bounds__(256) void k_gemm1(const bf16* __restrict__ P,
                                               const float* __restrict__ invn,
                                               float* __restrict__ scoreT, int b) {
  const int tid = threadIdx.x;
  const int wave = tid >> 5, lane = tid & 31;
  const int lo = lane & 15, hi = lane >> 4;
  const int l_base = (blockIdx.y * 4 + (wave >> 1)) * 32;   // 128 l-jobs
  const int p_base = (blockIdx.x * 2 + (wave & 1)) * 32;    // 128 p-jobs
  const bf16* Pb = P + (size_t)b * LL * KK;
  const bf16* arow0 = Pb + (size_t)(l_base + lo) * KK;
  const bf16* arow1 = arow0 + (size_t)16 * KK;
  const bf16* brow0 = Pb + (size_t)(p_base + lo) * KK;
  const bf16* brow1 = brow0 + (size_t)16 * KK;
  const int boff = hi ? 16 : 0;

  f32x8 acc00, acc01, acc10, acc11;
  for (int i = 0; i < 8; ++i) { acc00[i] = 0.f; acc01[i] = 0.f; acc10[i] = 0.f; acc11[i] = 0.f; }

  for (int k0 = 0; k0 < KK; k0 += 32) {
    if (k0 + 64 < KK) {
      __builtin_prefetch(arow0 + k0 + 64, 0, 1);
      __builtin_prefetch(arow1 + k0 + 64, 0, 1);
      __builtin_prefetch(brow0 + k0 + 64, 0, 1);
      __builtin_prefetch(brow1 + k0 + 64, 0, 1);
    }
    bf16x16 a0 = load_a_frag(arow0, k0, hi);
    bf16x16 a1 = load_a_frag(arow1, k0, hi);
    bf16x16 b0 = *(const bf16x16*)(brow0 + k0 + boff);
    bf16x16 b1 = *(const bf16x16*)(brow1 + k0 + boff);
    acc00 = WMMA_BF16(a0, b0, acc00);
    acc01 = WMMA_BF16(a0, b1, acc01);
    acc10 = WMMA_BF16(a1, b0, acc10);
    acc11 = WMMA_BF16(a1, b1, acc11);
  }

  const float* inv = invn + b * LL;
  f32x8 accs[2][2] = {{acc00, acc01}, {acc10, acc11}};
#pragma unroll
  for (int ti = 0; ti < 2; ++ti) {
#pragma unroll
    for (int tj = 0; tj < 2; ++tj) {
      const int pg = p_base + tj * 16 + lo;
#pragma unroll
      for (int r = 0; r < 8; ++r) {
        int lg = l_base + ti * 16 + r + (hi ? 8 : 0);
        scoreT[(size_t)pg * LL + lg] = accs[ti][tj][r] * inv[lg];
      }
    }
  }
}

// ---------------------------------------------------------------- K5: softmax over l (rows of scoreT), scale 10, clip 1e-8
__global__ __launch_bounds__(256) void k_softmax(const float* __restrict__ scoreT,
                                                 bf16* __restrict__ attnT) {
  const int p = blockIdx.x;
  const float* row = scoreT + (size_t)p * LL;
  bf16* orow = attnT + (size_t)p * LL;
  __shared__ float red[256];
  float m = -3.4e38f;
  for (int l = threadIdx.x; l < LL; l += 256) m = fmaxf(m, row[l] * 10.0f);
  red[threadIdx.x] = m; __syncthreads();
  for (int s = 128; s > 0; s >>= 1) {
    if (threadIdx.x < s) red[threadIdx.x] = fmaxf(red[threadIdx.x], red[threadIdx.x + s]);
    __syncthreads();
  }
  m = red[0]; __syncthreads();
  float ssum = 0.f;
  for (int l = threadIdx.x; l < LL; l += 256) ssum += __expf(row[l] * 10.0f - m);
  red[threadIdx.x] = ssum; __syncthreads();
  for (int s = 128; s > 0; s >>= 1) {
    if (threadIdx.x < s) red[threadIdx.x] += red[threadIdx.x + s];
    __syncthreads();
  }
  float inv = 1.0f / red[0];
  for (int l = threadIdx.x; l < LL; l += 256) {
    float v = __expf(row[l] * 10.0f - m) * inv;
    orow[l] = (bf16)fmaxf(v, 1e-8f);
  }
}

// ---------------------------------------------------------------- K6: transposed-conv as 4 accumulated GEMMs per parity class
// Register-blocked: each wave computes 32(ch) x 32(px) = 2x2 WMMA tiles, 4 combos accumulated.
// OOB attention rows are redirected to a zeroed dummy row -> no selects in the K-loop.
__global__ __launch_bounds__(256) void k_gemm2(const bf16* __restrict__ Bg,
                                               const bf16* __restrict__ attnT,
                                               const bf16* __restrict__ zrow,
                                               float* __restrict__ out, int b) {
  const int tid = threadIdx.x;
  const int wave = tid >> 5, lane = tid & 31;
  const int lo = lane & 15, hi = lane >> 4;
  const int cls = blockIdx.z;               // parity class (ry, rx)
  const int ry = cls >> 1, rx = cls & 1;
  const int cj = blockIdx.y * 2 + (wave >> 2);     // 0..5  -> 192 channels / 32
  const int nj = blockIdx.x * 4 + (wave & 3);      // 0..127 -> 4096 class pixels / 32
  const int c_base = cj * 32;
  const int qy = nj >> 1;
  const int qx0 = (nj & 1) * 32;

  int khs[2], dys[2], kws[2], dxs[2];
  if (ry == 0) { khs[0] = 1; khs[1] = 3; dys[0] = 0; dys[1] = -1; }
  else         { khs[0] = 0; khs[1] = 2; dys[0] = 1; dys[1] = 0;  }
  if (rx == 0) { kws[0] = 1; kws[1] = 3; dxs[0] = 0; dxs[1] = -1; }
  else         { kws[0] = 0; kws[1] = 2; dxs[0] = 1; dxs[1] = 0;  }

  const bf16* BgB = Bg + (size_t)b * 16 * CCH * LL;
  const bf16* arows0[4];
  const bf16* arows1[4];
  const bf16* brows[4][2];
#pragma unroll
  for (int j = 0; j < 4; ++j) {
    int kh = khs[j >> 1], kw = kws[j & 1];
    int dy = dys[j >> 1], dx = dxs[j & 1];
    int combo = kh * 4 + kw;
    arows0[j] = BgB + ((size_t)combo * CCH + c_base + lo) * LL;
    arows1[j] = arows0[j] + (size_t)16 * LL;
    int iy = qy + dy;
#pragma unroll
    for (int t = 0; t < 2; ++t) {
      int ix = qx0 + t * 16 + lo + dx;
      bool valid = ((unsigned)iy < 64u) && ((unsigned)ix < 64u);
      brows[j][t] = valid ? (attnT + (size_t)(iy * 64 + ix) * LL) : zrow;
    }
  }
  const int boff = hi ? 16 : 0;
  f32x8 acc00, acc01, acc10, acc11;
  for (int i = 0; i < 8; ++i) { acc00[i] = 0.f; acc01[i] = 0.f; acc10[i] = 0.f; acc11[i] = 0.f; }

  for (int l0 = 0; l0 < LL; l0 += 32) {
#pragma unroll
    for (int j = 0; j < 4; ++j) {
      if (l0 + 64 < LL) {
        __builtin_prefetch(arows0[j] + l0 + 64, 0, 1);
        __builtin_prefetch(arows1[j] + l0 + 64, 0, 1);
      }
      bf16x16 a0 = load_a_frag(arows0[j], l0, hi);
      bf16x16 a1 = load_a_frag(arows1[j], l0, hi);
      bf16x16 b0 = *(const bf16x16*)(brows[j][0] + l0 + boff);
      bf16x16 b1 = *(const bf16x16*)(brows[j][1] + l0 + boff);
      acc00 = WMMA_BF16(a0, b0, acc00);
      acc01 = WMMA_BF16(a0, b1, acc01);
      acc10 = WMMA_BF16(a1, b0, acc10);
      acc11 = WMMA_BF16(a1, b1, acc11);
    }
  }

  const int yo = 2 * qy + ry;
  f32x8 accs[2][2] = {{acc00, acc01}, {acc10, acc11}};
#pragma unroll
  for (int tc = 0; tc < 2; ++tc) {
#pragma unroll
    for (int tn = 0; tn < 2; ++tn) {
      const int xo = 2 * (qx0 + tn * 16 + lo) + rx;
#pragma unroll
      for (int r = 0; r < 8; ++r) {
        int c = c_base + tc * 16 + r + (hi ? 8 : 0);
        out[(((size_t)b * CCH + c) * HH + yo) * HH + xo] = accs[tc][tn][r] * 0.25f;
      }
    }
  }
}

// ----------------------------------------------------------------
extern "C" void kernel_launch(void* const* d_in, const int* in_sizes, int n_in,
                              void* d_out, int out_size, void* d_ws, size_t ws_size,
                              hipStream_t stream) {
  const float* bg = (const float*)d_in[0];   // background [4,192,128,128]
  const float* fg = (const float*)d_in[1];   // foreground [4,192,128,128]
  float* out = (float*)d_out;                // [4,192,128,128]
  char* ws = (char*)d_ws;
  (void)in_sizes; (void)n_in; (void)out_size; (void)ws_size;

  size_t o = 0;
  auto alignup = [](size_t v) { return (v + 255) & ~(size_t)255; };
  float* fgs    = (float*)(ws + o); o = alignup(o + sizeof(float) * (size_t)BATCH * CCH * hh * hh);
  bf16*  P      = (bf16*)(ws + o);  o = alignup(o + sizeof(bf16) * (size_t)BATCH * LL * KK);
  float* invn   = (float*)(ws + o); o = alignup(o + sizeof(float) * (size_t)BATCH * LL);
  bf16*  Bg     = (bf16*)(ws + o);  o = alignup(o + sizeof(bf16) * (size_t)BATCH * 16 * CCH * LL);
  bf16*  zrow   = (bf16*)(ws + o);  o = alignup(o + sizeof(bf16) * (size_t)LL);
  float* scoreT = (float*)(ws + o); o = alignup(o + sizeof(float) * (size_t)LL * LL);   // per-sample, reused
  bf16*  attnT  = (bf16*)(ws + o);  o = alignup(o + sizeof(bf16) * (size_t)LL * LL);    // per-sample, reused

  k_resize<<<dim3((BATCH * CCH * hh * hh + 255) / 256), 256, 0, stream>>>(fg, fgs);
  k_patch<<<dim3(LL, BATCH), 256, 0, stream>>>(fgs, P, invn);
  {
    size_t tot = (size_t)BATCH * 16 * CCH * LL;
    k_bgpack<<<dim3((unsigned)((tot + 255) / 256)), 256, 0, stream>>>(bg, Bg);
  }
  k_zerorow<<<dim3(LL / 256), 256, 0, stream>>>(zrow);
  for (int b = 0; b < BATCH; ++b) {
    k_gemm1<<<dim3(64, 32), 256, 0, stream>>>(P, invn, scoreT, b);       // 128x128 jobs of 32x32
    k_softmax<<<dim3(LL), 256, 0, stream>>>(scoreT, attnT);
    k_gemm2<<<dim3(32, 3, 4), 256, 0, stream>>>(Bg, attnT, zrow, out, b);  // 4 parity classes
  }
}
